// VQModel_83485574299926
// MI455X (gfx1250) — matrix operand order repaired and checked by reading, not compiled, not run on hardware
//
#include <hip/hip_runtime.h>

typedef __attribute__((ext_vector_type(16))) _Float16 v16h;
typedef __attribute__((ext_vector_type(8)))  float    v8f;

// ===========================================================================
// Prep: pack weights/codebook to f16 once (removes f32->f16 cvt from hot
// loops, zero-pads K and Cout so fragment builds are branch-free and the
// 32 B/lane weight reads vectorize to b128).
// ===========================================================================

// conv weight (COUT,CIN,4,4) f32 -> (COUT, KPAD) f16, zero-padded K
__global__ void pack_conv_w(const float* __restrict__ w,
                            _Float16* __restrict__ wp, int COUT, int CIN,
                            int KPAD) {
  const int i = blockIdx.x * 256 + threadIdx.x;
  if (i >= COUT * KPAD) return;
  const int co = i / KPAD, k = i % KPAD;
  float v = 0.f;
  if (k < CIN * 16) v = w[co * CIN * 16 + k];
  wp[i] = (_Float16)v;
}

// deconv weight (CIN,COUT,4,4) f32 -> [py][px][COP][CIN*4] f16 with the
// (ky,kx) taps pre-gathered per output parity; co zero-padded to COP.
__global__ void pack_deconv_w(const float* __restrict__ w,
                              _Float16* __restrict__ wp, int CIN, int COUT,
                              int COP) {
  const int K = CIN * 4;
  const int i = blockIdx.x * 256 + threadIdx.x;
  if (i >= 4 * COP * K) return;
  const int k = i % K;
  const int co = (i / K) % COP;
  const int pp = i / (K * COP);
  const int py = pp >> 1, px = pp & 1;
  const int ci = k >> 2, ty = (k >> 1) & 1, tx = k & 1;
  const int ky = (3 - py) - 2 * ty, kx = (3 - px) - 2 * tx;
  float v = 0.f;
  if (co < COUT) v = w[((ci * COUT + co) * 4 + ky) * 4 + kx];
  wp[i] = (_Float16)v;
}

// codebook (512,64) f32 -> f16, plus |c|^2 per code
__global__ void pack_cb(const float* __restrict__ cb,
                        _Float16* __restrict__ cbh, float* __restrict__ cc) {
  const int i = blockIdx.x * 256 + threadIdx.x;
  if (i < 512 * 64) cbh[i] = (_Float16)cb[i];
  if (i < 512) {
    float s = 0.f;
    for (int c = 0; c < 64; ++c) {
      const float v = cb[i * 64 + c];
      s += v * v;
    }
    cc[i] = s;
  }
}

// ===========================================================================
// Conv2d k=4 s=2 p=1 + bias + ReLU, implicit GEMM on v_wmma_f32_16x16x32_f16.
// One wave: 32 output pixels x 16 output channels (two independent WMMA
// accumulator chains sharing one B fragment).
// blockDim = 32*(COUT/16); grid = N * HOUT * (WOUT/32)
// ===========================================================================
template <typename Tin, int CIN, int COUT, int HIN, int WIN>
__global__ __launch_bounds__((COUT / 16) * 32) void conv4x4s2_wmma(
    const Tin* __restrict__ in, const _Float16* __restrict__ wp,
    const float* __restrict__ bias, _Float16* __restrict__ out) {
  constexpr int HOUT = HIN / 2, WOUT = WIN / 2;
  constexpr int XT = WOUT / 32;
  constexpr int K = CIN * 16;
  constexpr int CHUNKS = (K + 31) / 32;
  constexpr int KPAD = CHUNKS * 32;
  constexpr int CINP = KPAD / 16;  // padded input-channel rows in LDS
  constexpr int THREADS = (COUT / 16) * 32;

  const int b = blockIdx.x;
  const int xt = b % XT;
  const int oy = (b / XT) % HOUT;
  const int n = b / (XT * HOUT);
  const int ox0 = xt * 32;
  const int iy0 = 2 * oy - 1;
  const int ix0 = 2 * ox0 - 1;

  // staged input patch: CINP x 4 rows x 66 cols (stride 68), zero-padded
  __shared__ _Float16 s_in[CINP][4][68];

  const int tid = threadIdx.x;
  for (int e = tid; e < CINP * 4 * 68; e += THREADS) {
    const int dx = e % 68;
    const int dy = (e / 68) & 3;
    const int ci = e / (68 * 4);
    float v = 0.f;
    const int iy = iy0 + dy, ix = ix0 + dx;
    if (ci < CIN && dx < 66 && iy >= 0 && iy < HIN && ix >= 0 && ix < WIN)
      v = (float)in[((size_t)(n * CIN + ci) * HIN + iy) * WIN + ix];
    s_in[ci][dy][dx] = (_Float16)v;
  }
  __syncthreads();

  const int lane = tid & 31;
  const int wv = tid >> 5;
  const int lh = lane >> 4;
  const int lr = lane & 15;
  const int co0 = wv * 16;

  v8f acc0 = {}, acc1 = {};
  for (int kc = 0; kc < CHUNKS; ++kc) {
    v16h a0, a1, bm;
#pragma unroll
    for (int s = 0; s < 16; ++s) {
      // A fragment, contiguous k order used here (matches B): slot s -> K = lh*16 + s
      const int ks = lh * 16 + s;
      const int k = kc * 32 + ks;
      const int ci = k >> 4, tap = k & 15;
      const int ky = tap >> 2, kx = tap & 3;
      a0[s] = s_in[ci][ky][2 * lr + kx];
      a1[s] = s_in[ci][ky][2 * lr + 32 + kx];
    }
#pragma unroll
    for (int s = 0; s < 16; ++s) {
      // B fragment, contiguous k order used here (matches A): slot s -> K = lh*16 + s, column = lr
      const int k = kc * 32 + lh * 16 + s;
      bm[s] = wp[(size_t)(co0 + lr) * KPAD + k];
    }
    acc0 = __builtin_amdgcn_wmma_f32_16x16x32_f16(false, a0, false, bm,
                                                  (short)0, acc0, false, false);
    acc1 = __builtin_amdgcn_wmma_f32_16x16x32_f16(false, a1, false, bm,
                                                  (short)0, acc1, false, false);
  }

  const int co = co0 + lr;
  const float bb = bias[co];
  _Float16* obase =
      out + ((size_t)(n * COUT + co) * HOUT + oy) * WOUT + ox0;
#pragma unroll
  for (int v = 0; v < 8; ++v) {
    const int m = v + 8 * lh;  // C layout: lanes16-31 hold rows 8..15
    float v0 = acc0[v] + bb, v1 = acc1[v] + bb;
    v0 = v0 > 0.f ? v0 : 0.f;
    v1 = v1 > 0.f ? v1 : 0.f;
    obase[m] = (_Float16)v0;
    obase[16 + m] = (_Float16)v1;
  }
}

// ===========================================================================
// ConvTranspose2d k=4 s=2 p=1 as a gather-GEMM over same-parity output
// columns (fixed 2x2 tap pattern => K = CIN*4). One wave: 32 same-parity
// output columns x 16 output channels, dual WMMA chains.
// blockDim = 32*NCO16; grid = N * HOUT * (WOUT/64) * 2 parities
// ===========================================================================
template <int CIN, int COUT, int NCO16, int HIN, int WIN, bool SIG,
          typename Tout>
__global__ __launch_bounds__(NCO16 * 32) void deconv4x4s2_wmma(
    const _Float16* __restrict__ in, const _Float16* __restrict__ wp,
    const float* __restrict__ bias, Tout* __restrict__ out) {
  constexpr int HOUT = HIN * 2, WOUT = WIN * 2;
  constexpr int XT = WOUT / 64;
  constexpr int K = CIN * 4;
  constexpr int CHUNKS = K / 32;
  constexpr int COP = NCO16 * 16;
  constexpr int THREADS = NCO16 * 32;

  const int b = blockIdx.x;
  const int p = b & 1;  // output-column parity
  const int xt = (b >> 1) % XT;
  const int oy = ((b >> 1) / XT) % HOUT;
  const int n = (b >> 1) / (XT * HOUT);

  const int py = oy & 1;
  // contributing input rows: iy = iyb + ty (ty = 0,1)
  const int iyb = ((oy + 1) >> 1) - 1;
  // contributing input cols for ox = 64*xt + 2*mm + p: ix = ixL + mm + tx
  const int ixL = 32 * xt - 1 + p;

  __shared__ _Float16 s_in[CIN][2][34];
  const int tid = threadIdx.x;
  for (int e = tid; e < CIN * 2 * 34; e += THREADS) {
    const int dx = e % 34;
    const int ty = (e / 34) & 1;
    const int ci = e / 68;
    float v = 0.f;
    const int iy = iyb + ty, ix = ixL + dx;
    if (iy >= 0 && iy < HIN && ix >= 0 && ix < WIN)
      v = (float)in[((size_t)(n * CIN + ci) * HIN + iy) * WIN + ix];
    s_in[ci][ty][dx] = (_Float16)v;
  }
  __syncthreads();

  const int lane = tid & 31;
  const int wvid = tid >> 5;
  const int lh = lane >> 4;
  const int lr = lane & 15;
  const int co0 = wvid * 16;
  // pre-gathered per-parity weights: wp[((py*2+px)*COP + co)*K + k]
  const _Float16* wrow = wp + ((size_t)(py * 2 + p) * COP + co0 + lr) * K;

  v8f acc0 = {}, acc1 = {};
  for (int kc = 0; kc < CHUNKS; ++kc) {
    v16h a0, a1, bm;
#pragma unroll
    for (int s = 0; s < 16; ++s) {
      const int ks = lh * 16 + s;
      const int k = kc * 32 + ks;  // k = ci*4 + ty*2 + tx
      const int ci = k >> 2, ty = (k >> 1) & 1, tx = k & 1;
      a0[s] = s_in[ci][ty][lr + tx];
      a1[s] = s_in[ci][ty][lr + 16 + tx];
    }
#pragma unroll
    for (int s = 0; s < 16; ++s) {
      bm[s] = wrow[kc * 32 + lh * 16 + s];
    }
    acc0 = __builtin_amdgcn_wmma_f32_16x16x32_f16(false, a0, false, bm,
                                                  (short)0, acc0, false, false);
    acc1 = __builtin_amdgcn_wmma_f32_16x16x32_f16(false, a1, false, bm,
                                                  (short)0, acc1, false, false);
  }

  const int co = co0 + lr;
  if (co < COUT) {
    const float bb = bias[co];
    Tout* obase = out + ((size_t)(n * COUT + co) * HOUT + oy) * WOUT;
#pragma unroll
    for (int v = 0; v < 8; ++v) {
      const int m = v + 8 * lh;
      float v0 = acc0[v] + bb, v1 = acc1[v] + bb;
      if (SIG) {
        v0 = 1.f / (1.f + __expf(-v0));
        v1 = 1.f / (1.f + __expf(-v1));
      } else {
        v0 = v0 > 0.f ? v0 : 0.f;
        v1 = v1 > 0.f ? v1 : 0.f;
      }
      obase[64 * xt + 2 * m + p] = (Tout)v0;
      obase[64 * xt + 2 * (16 + m) + p] = (Tout)v1;
    }
  }
}

// ===========================================================================
// Vector quantization: per position argmin over 512 codes of (|c|^2 - 2 x.c).
// One block = one 32-position row; 4 waves sweep the 32 code tiles with dual
// WMMA chains. Input row staged into LDS via the CDNA5 async DMA path
// (GLOBAL_LOAD_ASYNC_TO_LDS_B128, ASYNCcnt) -- perfectly aligned, no padding.
// ===========================================================================
__global__ __launch_bounds__(128) void vq_wmma(
    const _Float16* __restrict__ h, const float* __restrict__ cb,
    const _Float16* __restrict__ cbh, const float* __restrict__ cc,
    _Float16* __restrict__ q, float* __restrict__ partial) {
  const int b = blockIdx.x;  // 2048 blocks: 64 n * 32 y
  const int y = b & 31;
  const int n = b >> 5;

  __shared__ _Float16 s_a[64][32];  // [ci][x], 4 KB, rows = 64 B
  __shared__ float s_bv[4][32];
  __shared__ int s_bi[4][32];
  __shared__ int s_idx[32];
  __shared__ float s_ws[4];

  const int tid = threadIdx.x;
  {
    // 2048 f16 = 4 KB: 128 lanes x 2 async b128 copies (16 B each).
    // e = tid*8 keeps each 16 B transfer inside one 64 B row of h.
    const int e = tid * 8;
    const int ci = e >> 5, m = e & 31;
    const _Float16* g0 = h + ((size_t)(n * 64 + ci) * 32 + y) * 32 + m;
    const _Float16* g1 = h + ((size_t)(n * 64 + ci + 32) * 32 + y) * 32 + m;
    // low 32 bits of a generic LDS pointer are the LDS byte offset
    const unsigned l0 = (unsigned)(size_t)&s_a[ci][m];
    const unsigned l1 = (unsigned)(size_t)&s_a[ci + 32][m];
    asm volatile("global_load_async_to_lds_b128 %0, %1, off" ::"v"(l0),
                 "v"(g0)
                 : "memory");
    asm volatile("global_load_async_to_lds_b128 %0, %1, off" ::"v"(l1),
                 "v"(g1)
                 : "memory");
    asm volatile("s_wait_asynccnt 0x0" ::: "memory");
  }
  __syncthreads();

  const int lane = tid & 31, wv = tid >> 5, lh = lane >> 4, lr = lane & 15;

  float bv0[8], bv1[8];
  int bi0[8], bi1[8];
#pragma unroll
  for (int v = 0; v < 8; ++v) {
    bv0[v] = bv1[v] = 3.4e38f;
    bi0[v] = bi1[v] = 0x7fffffff;
  }

  for (int j = 0; j < 8; ++j) {  // 4 waves x 8 tiles = 32 code tiles
    const int code0 = (wv + 4 * j) * 16;
    v8f acc0 = {}, acc1 = {};
#pragma unroll
    for (int kc = 0; kc < 2; ++kc) {
      v16h a0, a1, bm;
#pragma unroll
      for (int s = 0; s < 16; ++s) {
        const int ks = lh * 16 + s;
        a0[s] = s_a[kc * 32 + ks][lr];
        a1[s] = s_a[kc * 32 + ks][lr + 16];
        bm[s] = cbh[(size_t)(code0 + lr) * 64 + kc * 32 + lh * 16 + s];
      }
      acc0 = __builtin_amdgcn_wmma_f32_16x16x32_f16(false, a0, false, bm,
                                                    (short)0, acc0, false,
                                                    false);
      acc1 = __builtin_amdgcn_wmma_f32_16x16x32_f16(false, a1, false, bm,
                                                    (short)0, acc1, false,
                                                    false);
    }
    const int code = code0 + lr;
    const float c2 = cc[code];
#pragma unroll
    for (int v = 0; v < 8; ++v) {
      const float s0 = c2 - 2.f * acc0[v];
      const float s1 = c2 - 2.f * acc1[v];
      if (s0 < bv0[v] || (s0 == bv0[v] && code < bi0[v])) {
        bv0[v] = s0;
        bi0[v] = code;
      }
      if (s1 < bv1[v] || (s1 == bv1[v] && code < bi1[v])) {
        bv1[v] = s1;
        bi1[v] = code;
      }
    }
  }

  // reduce argmin across the 16 lanes holding each row
#pragma unroll
  for (int v = 0; v < 8; ++v) {
    float a = bv0[v], c = bv1[v];
    int ai = bi0[v], ci2 = bi1[v];
    for (int mask = 8; mask >= 1; mask >>= 1) {
      const float oa = __shfl_xor(a, mask, 16);
      const int oai = __shfl_xor(ai, mask, 16);
      if (oa < a || (oa == a && oai < ai)) {
        a = oa;
        ai = oai;
      }
      const float oc = __shfl_xor(c, mask, 16);
      const int oci = __shfl_xor(ci2, mask, 16);
      if (oc < c || (oc == c && oci < ci2)) {
        c = oc;
        ci2 = oci;
      }
    }
    if (lr == 0) {
      const int row = v + 8 * lh;
      s_bv[wv][row] = a;
      s_bi[wv][row] = ai;
      s_bv[wv][16 + row] = c;
      s_bi[wv][16 + row] = ci2;
    }
  }
  __syncthreads();
  if (tid < 32) {
    float bv = s_bv[0][tid];
    int bi = s_bi[0][tid];
    for (int wq = 1; wq < 4; ++wq) {
      const float ov = s_bv[wq][tid];
      const int oi = s_bi[wq][tid];
      if (ov < bv || (ov == bv && oi < bi)) {
        bv = ov;
        bi = oi;
      }
    }
    s_idx[tid] = bi;
  }
  __syncthreads();

  // emit q (NCHW, f16) and accumulate (h - q)^2
  float lsum = 0.f;
  for (int e = tid; e < 2048; e += 128) {
    const int m = e & 31, ci = e >> 5;
    const float qv = cb[(size_t)s_idx[m] * 64 + ci];
    const float hv = (float)s_a[ci][m];
    const float d = hv - qv;
    lsum += d * d;
    q[((size_t)(n * 64 + ci) * 32 + y) * 32 + m] = (_Float16)qv;
  }
  for (int mask = 16; mask >= 1; mask >>= 1) lsum += __shfl_xor(lsum, mask, 32);
  if (lane == 0) s_ws[wv] = lsum;
  __syncthreads();
  if (tid == 0) partial[b] = s_ws[0] + s_ws[1] + s_ws[2] + s_ws[3];
}

// deterministic tree-reduction of block partials -> both loss scalars
__global__ void vq_finalize(const float* __restrict__ partial,
                            float* __restrict__ out, int nPartial,
                            long long off) {
  __shared__ float s[256];
  float sum = 0.f;
  for (int i = threadIdx.x; i < nPartial; i += 256) sum += partial[i];
  s[threadIdx.x] = sum;
  __syncthreads();
  for (int st = 128; st > 0; st >>= 1) {
    if ((int)threadIdx.x < st) s[threadIdx.x] += s[threadIdx.x + st];
    __syncthreads();
  }
  if (threadIdx.x == 0) {
    const float m = s[0] / 4194304.0f;  // mean over 64*64*32*32 elements
    out[off] = m;
    out[off + 1] = m;
  }
}

// ===========================================================================
extern "C" void kernel_launch(void* const* d_in, const int* in_sizes, int n_in,
                              void* d_out, int out_size, void* d_ws,
                              size_t ws_size, hipStream_t stream) {
  const float* x = (const float*)d_in[0];
  const float* w1 = (const float*)d_in[1];
  const float* b1 = (const float*)d_in[2];
  const float* w2 = (const float*)d_in[3];
  const float* b2 = (const float*)d_in[4];
  const float* w3 = (const float*)d_in[5];
  const float* b3 = (const float*)d_in[6];
  const float* cb = (const float*)d_in[7];
  const float* dw1 = (const float*)d_in[8];
  const float* db1 = (const float*)d_in[9];
  const float* dw2 = (const float*)d_in[10];
  const float* db2 = (const float*)d_in[11];
  const float* dw3 = (const float*)d_in[12];
  const float* db3 = (const float*)d_in[13];
  float* out = (float*)d_out;

  // workspace layout (f16 activations halve HBM traffic vs f32)
  _Float16* h1 = (_Float16*)d_ws;               // 64*32*128*128
  _Float16* h2 = h1 + (size_t)33554432;         // 64*64*64*64
  _Float16* h3 = h2 + (size_t)16777216;         // 64*64*32*32
  _Float16* qb = h3 + (size_t)4194304;          // 64*64*32*32
  _Float16* g1 = qb + (size_t)4194304;          // 64*64*64*64
  _Float16* g2 = g1 + (size_t)16777216;         // 64*32*128*128
  _Float16* w1p = g2 + (size_t)33554432;        // 32*64
  _Float16* w2p = w1p + 32 * 64;                // 64*512
  _Float16* w3p = w2p + 64 * 512;               // 64*1024
  _Float16* d1p = w3p + 64 * 1024;              // 4*64*256
  _Float16* d2p = d1p + 4 * 64 * 256;           // 4*32*256
  _Float16* d3p = d2p + 4 * 32 * 256;           // 4*16*128
  _Float16* cbh = d3p + 4 * 16 * 128;           // 512*64
  float* cc = (float*)(cbh + 512 * 64);         // 512
  float* partial = cc + 512;                    // 2048

  // ---- pack params to f16 (tiny, once per call) ----
  pack_conv_w<<<(32 * 64 + 255) / 256, 256, 0, stream>>>(w1, w1p, 32, 3, 64);
  pack_conv_w<<<(64 * 512 + 255) / 256, 256, 0, stream>>>(w2, w2p, 64, 32, 512);
  pack_conv_w<<<(64 * 1024 + 255) / 256, 256, 0, stream>>>(w3, w3p, 64, 64,
                                                           1024);
  pack_deconv_w<<<(4 * 64 * 256 + 255) / 256, 256, 0, stream>>>(dw1, d1p, 64,
                                                                64, 64);
  pack_deconv_w<<<(4 * 32 * 256 + 255) / 256, 256, 0, stream>>>(dw2, d2p, 64,
                                                                32, 32);
  pack_deconv_w<<<(4 * 16 * 128 + 255) / 256, 256, 0, stream>>>(dw3, d3p, 32, 3,
                                                                16);
  pack_cb<<<(512 * 64 + 255) / 256, 256, 0, stream>>>(cb, cbh, cc);

  // ---- encoder ----
  conv4x4s2_wmma<float, 3, 32, 256, 256>
      <<<64 * 128 * 4, 64, 0, stream>>>(x, w1p, b1, h1);
  conv4x4s2_wmma<_Float16, 32, 64, 128, 128>
      <<<64 * 64 * 2, 128, 0, stream>>>(h1, w2p, b2, h2);
  conv4x4s2_wmma<_Float16, 64, 64, 64, 64>
      <<<64 * 32 * 1, 128, 0, stream>>>(h2, w3p, b3, h3);

  // ---- vector quantization ----
  vq_wmma<<<2048, 128, 0, stream>>>(h3, cb, cbh, cc, qb, partial);

  // ---- decoder (ste output == q) ----
  deconv4x4s2_wmma<64, 64, 4, 32, 32, false, _Float16>
      <<<64 * 64 * 1 * 2, 128, 0, stream>>>(qb, d1p, db1, g1);
  deconv4x4s2_wmma<64, 32, 2, 64, 64, false, _Float16>
      <<<64 * 128 * 2 * 2, 64, 0, stream>>>(g1, d2p, db2, g2);
  deconv4x4s2_wmma<32, 3, 1, 128, 128, true, float>
      <<<64 * 256 * 4 * 2, 32, 0, stream>>>(g2, d3p, db3, out);

  // ---- losses (deterministic reduction) ----
  vq_finalize<<<1, 256, 0, stream>>>(partial, out, 2048, (long long)12582912);
}
